// DGCN_83674552860914
// MI455X (gfx1250) — matrix operand order repaired and checked by reading, not compiled, not run on hardware
//
#include <hip/hip_runtime.h>
#include <math.h>

typedef __attribute__((ext_vector_type(16))) _Float16 v16h;
typedef __attribute__((ext_vector_type(8)))  float    v8f;

#define N_NODES 50000
#define F_DIM   64

// ---------------------------------------------------------------------------
// Degree / normalization kernels
// ---------------------------------------------------------------------------
__global__ void deg_init(float* d0, float* d1, float* d2, int n) {
    int i = blockIdx.x * blockDim.x + threadIdx.x;
    if (i < n) { d0[i] = 1.0f; d1[i] = 1.0f; d2[i] = 1.0f; }   // self-loop weight 1
}

__global__ void deg_accum(const int* __restrict__ c0, const int* __restrict__ c1,
                          const int* __restrict__ c2,
                          const float* __restrict__ w1, const float* __restrict__ w2,
                          float* d0, float* d1, float* d2, int nE) {
    int e = blockIdx.x * blockDim.x + threadIdx.x;
    if (e >= nE) return;
    atomicAdd(d0 + c0[e], 1.0f);
    atomicAdd(d1 + c1[e], w1[e]);
    atomicAdd(d2 + c2[e], w2[e]);
}

__global__ void deg_to_dis(float* d0, float* d1, float* d2, int n) {
    int i = blockIdx.x * blockDim.x + threadIdx.x;
    if (i >= n) return;
    d0[i] = rsqrtf(d0[i]);
    d1[i] = rsqrtf(d1[i]);
    d2[i] = rsqrtf(d2[i]);
}

// ---------------------------------------------------------------------------
// WMMA GEMM:  OUT[nrows x NCOLS] = X[nrows x K] * W[NCOLS x K]^T (+bias)
// One 16x16 output tile strip per wave; W staged in LDS as f16 per block.
// nrows must be a multiple of 16 (50000 = 16*3125). K multiple of 32.
// ---------------------------------------------------------------------------
template<int NCOLS>
__global__ void gemm_xwt(const float* __restrict__ X, const float* __restrict__ W,
                         const float* __restrict__ bias, float* __restrict__ OUT,
                         int nrows, int K) {
    extern __shared__ _Float16 Wlds[];                 // NCOLS * K halves
    const int total = NCOLS * K;
    for (int i = threadIdx.x; i < total; i += blockDim.x)
        Wlds[i] = (_Float16)W[i];
    __syncthreads();

    const int lane  = threadIdx.x & 31;
    const int wave  = threadIdx.x >> 5;
    const int mtile = blockIdx.x * (blockDim.x >> 5) + wave;
    const int row0  = mtile * 16;
    if (row0 >= nrows) return;

    const int lrow = lane & 15;        // row (A) / col (B,D) within tile
    const int hi   = lane >> 4;        // upper half-wave selector

    constexpr int NT = NCOLS / 16;
    v8f zero = {};
    v8f acc[NT];
#pragma unroll
    for (int t = 0; t < NT; ++t) acc[t] = zero;

    const float* xrow = X + (size_t)(row0 + lrow) * K;

    for (int k0 = 0; k0 < K; k0 += 32) {
        // A fragment: a[i] = X[row0+lrow][k0 + (i<8 ? i : i+8) + 8*hi]
        v16h a;
        const float* ap = xrow + k0 + hi * 8;
#pragma unroll
        for (int i = 0; i < 8; ++i) a[i] = (_Float16)ap[i];
#pragma unroll
        for (int i = 0; i < 8; ++i) a[8 + i] = (_Float16)ap[16 + i];

#pragma unroll
        for (int t = 0; t < NT; ++t) {
            // B fragment: b[i] = W[t*16+lrow][k0 + (i<8 ? i : i+8) + 8*hi]   (B = W^T)
            const _Float16* bp = Wlds + (size_t)(t * 16 + lrow) * K + k0 + hi * 8;
            v16h b;
#pragma unroll
            for (int i = 0; i < 8; ++i) b[i] = bp[i];
#pragma unroll
            for (int i = 0; i < 8; ++i) b[8 + i] = bp[16 + i];
            acc[t] = __builtin_amdgcn_wmma_f32_16x16x32_f16(
                false, a, false, b, (short)0, acc[t], false, false);
        }
    }

#pragma unroll
    for (int t = 0; t < NT; ++t) {
        const int cb  = t * 16 + lrow;
        const float bb = bias ? bias[cb] : 0.0f;
#pragma unroll
        for (int i = 0; i < 8; ++i) {
            const int r = row0 + i + hi * 8;           // D layout (ISA 7.12.2)
            OUT[(size_t)r * NCOLS + cb] = acc[t][i] + bb;
        }
    }
}

// ---------------------------------------------------------------------------
// Self-loop contribution initializes hcat:  hcat[n][s*64+f] = dis_s[n]^2 * h[n][f]
// ---------------------------------------------------------------------------
__global__ void selfloop_init(const float* __restrict__ h,
                              const float* __restrict__ d0, const float* __restrict__ d1,
                              const float* __restrict__ d2,
                              float* __restrict__ hcat, int n) {
    int tid = blockIdx.x * blockDim.x + threadIdx.x;
    if (tid >= n * F_DIM) return;
    int node = tid >> 6, f = tid & 63;
    float hv = h[tid];
    float a0 = d0[node], a1 = d1[node], a2 = d2[node];
    float* o = hcat + (size_t)node * 192 + f;
    o[0]   = a0 * a0 * hv;
    o[64]  = a1 * a1 * hv;
    o[128] = a2 * a2 * hv;
}

// ---------------------------------------------------------------------------
// Edge scatter: out[col][col_off + f] += dis[row]*w*dis[col] * h[row][f]
// 16 lanes per edge, float4 gathers, f32 global atomics.
// ---------------------------------------------------------------------------
__global__ void edge_scatter(const int* __restrict__ rows, const int* __restrict__ cols,
                             const float* __restrict__ w, const float* __restrict__ dis,
                             const float* __restrict__ h, float* __restrict__ out,
                             int col_off, int nE) {
    int tid = blockIdx.x * blockDim.x + threadIdx.x;
    int e = tid >> 4;
    if (e >= nE) return;
    int sub = tid & 15;
    int r = rows[e], c = cols[e];
    float we = w ? w[e] : 1.0f;
    float norm = dis[r] * we * dis[c];
    const float4 hv = *(const float4*)(h + (size_t)r * F_DIM + sub * 4);
    float* op = out + (size_t)c * 192 + col_off + sub * 4;
    atomicAdd(op + 0, norm * hv.x);
    atomicAdd(op + 1, norm * hv.y);
    atomicAdd(op + 2, norm * hv.z);
    atomicAdd(op + 3, norm * hv.w);
}

// ---------------------------------------------------------------------------
// hcat = relu(hcat + bias[ j % 64 ])
// ---------------------------------------------------------------------------
__global__ void bias_relu(float* __restrict__ hcat, const float* __restrict__ bias, int n) {
    int tid = blockIdx.x * blockDim.x + threadIdx.x;
    if (tid >= n * 192) return;
    int j = tid % 192;
    float v = hcat[tid] + bias[j & 63];
    hcat[tid] = v > 0.0f ? v : 0.0f;
}

// ---------------------------------------------------------------------------
// In-place per-row log-softmax over 16 logits
// ---------------------------------------------------------------------------
__global__ void log_softmax16(float* __restrict__ logits, int n) {
    int i = blockIdx.x * blockDim.x + threadIdx.x;
    if (i >= n) return;
    float v[16];
    float m = -3.402823466e+38f;
    float* p = logits + (size_t)i * 16;
#pragma unroll
    for (int j = 0; j < 16; ++j) { v[j] = p[j]; m = fmaxf(m, v[j]); }
    float s = 0.0f;
#pragma unroll
    for (int j = 0; j < 16; ++j) s += __expf(v[j] - m);
    float ls = __logf(s);
#pragma unroll
    for (int j = 0; j < 16; ++j) p[j] = v[j] - m - ls;
}

// ---------------------------------------------------------------------------
extern "C" void kernel_launch(void* const* d_in, const int* in_sizes, int n_in,
                              void* d_out, int out_size, void* d_ws, size_t ws_size,
                              hipStream_t stream) {
    const float* x      = (const float*)d_in[0];
    const int*   ei     = (const int*)  d_in[1];
    const int*   e_in   = (const int*)  d_in[2];
    const int*   e_out  = (const int*)  d_in[3];
    const float* in_w   = (const float*)d_in[4];
    const float* out_w  = (const float*)d_in[5];
    const float* lin1_w = (const float*)d_in[6];
    const float* lin2_w = (const float*)d_in[7];
    const float* bias1  = (const float*)d_in[8];
    const float* bias2  = (const float*)d_in[9];
    const float* conv_w = (const float*)d_in[10];
    const float* conv_b = (const float*)d_in[11];
    float* outp = (float*)d_out;

    const int N = in_sizes[0] / 128;       // 50000
    const int E = in_sizes[4];             // 800000

    // edge_index layout: [2, E] flat
    const int* r0 = ei;          const int* c0 = ei    + E;
    const int* r1 = e_in;        const int* c1 = e_in  + E;
    const int* r2 = e_out;       const int* c2 = e_out + E;

    // Workspace layout (floats): dis0|dis1|dis2|h[N*64]|hcat[N*192]
    float* ws   = (float*)d_ws;
    float* dis0 = ws;
    float* dis1 = ws + N;
    float* dis2 = ws + 2 * N;
    float* h    = ws + 3 * N;
    float* hcat = h + (size_t)N * F_DIM;

    const int TB = 256;
    const int gN   = (N + TB - 1) / TB;
    const int gE   = (E + TB - 1) / TB;
    const int gE16 = (E * 16 + TB - 1) / TB;
    const int gNF  = (N * F_DIM + TB - 1) / TB;
    const int gNC  = (N * 192 + TB - 1) / TB;
    const int mtiles = N / 16;                       // 3125
    const int gGemm  = (mtiles + 7) / 8;             // 8 waves/block

    // --- normalization ---
    deg_init<<<gN, TB, 0, stream>>>(dis0, dis1, dis2, N);
    deg_accum<<<gE, TB, 0, stream>>>(c0, c1, c2, in_w, out_w, dis0, dis1, dis2, E);
    deg_to_dis<<<gN, TB, 0, stream>>>(dis0, dis1, dis2, N);

    // --- layer 1: h = x @ lin1^T ---
    gemm_xwt<64><<<gGemm, TB, 128 * 64 * sizeof(_Float16), stream>>>(x, lin1_w, nullptr, h, N, 128);
    selfloop_init<<<gNF, TB, 0, stream>>>(h, dis0, dis1, dis2, hcat, N);
    edge_scatter<<<gE16, TB, 0, stream>>>(r0, c0, nullptr, dis0, h, hcat, 0,   E);
    edge_scatter<<<gE16, TB, 0, stream>>>(r1, c1, in_w,    dis1, h, hcat, 64,  E);
    edge_scatter<<<gE16, TB, 0, stream>>>(r2, c2, out_w,   dis2, h, hcat, 128, E);
    bias_relu<<<gNC, TB, 0, stream>>>(hcat, bias1, N);

    // --- layer 2: h = hcat @ lin2^T ---
    gemm_xwt<64><<<gGemm, TB, 192 * 64 * sizeof(_Float16), stream>>>(hcat, lin2_w, nullptr, h, N, 192);
    selfloop_init<<<gNF, TB, 0, stream>>>(h, dis0, dis1, dis2, hcat, N);
    edge_scatter<<<gE16, TB, 0, stream>>>(r0, c0, nullptr, dis0, h, hcat, 0,   E);
    edge_scatter<<<gE16, TB, 0, stream>>>(r1, c1, in_w,    dis1, h, hcat, 64,  E);
    edge_scatter<<<gE16, TB, 0, stream>>>(r2, c2, out_w,   dis2, h, hcat, 128, E);
    bias_relu<<<gNC, TB, 0, stream>>>(hcat, bias2, N);

    // --- head: logits = hcat @ conv_w^T + conv_b ; log_softmax ---
    gemm_xwt<16><<<gGemm, TB, 192 * 16 * sizeof(_Float16), stream>>>(hcat, conv_w, conv_b, outp, N, 192);
    log_softmax16<<<gN, TB, 0, stream>>>(outp, N);
}
